// AveragedAdapter_91328184582706
// MI455X (gfx1250) — matrix operand order repaired and checked by reading, not compiled, not run on hardware
//
#include <hip/hip_runtime.h>
#include <hip/hip_bf16.h>
#include <math.h>

// ---------------------------------------------------------------------------
// AveragedAdapter on MI455X (gfx1250): 64 batched 2-layer MLPs + MSE reduce.
// Memory-bound on streaming 512 MB of f32 weights; compute via
// v_wmma_f32_16x16x32_bf16 (bf16 in / f32 accumulate).
// K-depth 64 staged per barrier; packed bf16 pair conversion + b32/b64 LDS
// stores on the weight/feature staging paths.
// ---------------------------------------------------------------------------

typedef __bf16 bf16_t;
typedef __attribute__((ext_vector_type(2)))  bf16_t v2bf;
typedef __attribute__((ext_vector_type(4)))  bf16_t v4bf;
typedef __attribute__((ext_vector_type(8)))  bf16_t v8bf;
typedef __attribute__((ext_vector_type(16))) bf16_t v16bf;
typedef __attribute__((ext_vector_type(8)))  float  v8f;

#define B_    128
#define E_    8
#define D_    512
#define H_    2048
#define BT    32      // batch rows per workgroup
#define HC    128     // H chunk
#define KC    32      // WMMA K step
#define KSTAGE 64     // K depth staged per barrier (2 WMMA K-steps)
#define LDA1  520     // features tile stride (conflict-free, rows 16B aligned)
#define LDA2  136     // h tile stride
#define LDT   72      // transposed weight tile stride (16B-aligned rows)
#define NTHREADS 256
#define SCALE (1.0f / (128.0f * 512.0f * 8.0f))

__device__ __forceinline__ v8f wmma_bf16(v16bf a, v16bf b, v8f c) {
  // D = A(16x32 bf16) x B(32x16 bf16) + C(16x16 f32)
  return __builtin_amdgcn_wmma_f32_16x16x32_bf16(
      /*neg_a=*/false, a, /*neg_b=*/false, b,
      /*c_mod=*/(short)0, c, /*reuse_a=*/false, /*reuse_b=*/false);
}

// A fragment (16x32, row-major K-contiguous tile in LDS).
// ISA: lanes 0-15 -> M=lane, K 0..7 (VGPR0-3), 16..23 (VGPR4-7);
//      lanes 16-31 -> same M, K 8..15 and 24..31.
__device__ __forceinline__ v16bf frag_A(const bf16_t* tile, int mBase, int kBase,
                                        int lda, int lane) {
  const int r = mBase + (lane & 15);
  const int k = kBase + ((lane >> 4) << 3);   // 0 or 8
  const bf16_t* p = tile + r * lda + k;
  v8bf lo = *(const v8bf*)p;          // K k..k+7
  v8bf hi = *(const v8bf*)(p + 16);   // K k+16..k+23
  return __builtin_shufflevector(lo, hi, 0,1,2,3,4,5,6,7,8,9,10,11,12,13,14,15);
}

// B fragment (32x16) from an N-major, K-contiguous (transposed) LDS tile.
// ISA: lanes 0-15 -> N=lane, K 0..7, 16..23; lanes 16-31 -> N=lane-16, K 8..15, 24..31.
__device__ __forceinline__ v16bf frag_B(const bf16_t* tileT, int nBase, int kBase,
                                        int ldb, int lane) {
  const int c = nBase + (lane & 15);
  const int k = kBase + ((lane >> 4) << 3);   // 0 or 8
  const bf16_t* p = tileT + c * ldb + k;
  v8bf lo = *(const v8bf*)p;
  v8bf hi = *(const v8bf*)(p + 16);
  return __builtin_shufflevector(lo, hi, 0,1,2,3,4,5,6,7,8,9,10,11,12,13,14,15);
}

// Convert a (k, k+1) f32 pair to packed bf16 and store as one 32-bit LDS word.
__device__ __forceinline__ void store_pk(bf16_t* p, float a, float b) {
  v2bf t;
  t.x = (bf16_t)a;
  t.y = (bf16_t)b;
  *(v2bf*)p = t;
}

__global__ void zero_out_kernel(float* out) { out[0] = 0.0f; }

__global__ __launch_bounds__(NTHREADS)
void adapter_mse_kernel(const float* __restrict__ features,
                        const float* __restrict__ target,
                        const float* __restrict__ W1,
                        const float* __restrict__ b1,
                        const float* __restrict__ W2,
                        const float* __restrict__ b2,
                        float* __restrict__ out) {
  __shared__ __align__(16) bf16_t featA[BT * LDA1];   // 33280 B  bf16 features
  __shared__ __align__(16) bf16_t btB1[HC * LDT];     // 18432 B  W1 K=64 chunk, T
  __shared__ __align__(16) bf16_t btB2[D_ * LDT];     // 73728 B  W2 K=64 chunk, T
  __shared__ __align__(16) bf16_t hA[BT * LDA2];      //  8704 B  gelu(h) tile
  __shared__ float red[NTHREADS / 32];

  const int tid  = threadIdx.x;
  const int lane = tid & 31;
  const int wave = tid >> 5;            // 0..7

  const int wg    = blockIdx.x;         // 0..255
  const int pair  = wg >> 2;            // 0..63
  const int a     = pair >> 3;
  const int e     = pair & (E_ - 1);
  const int bBase = (wg & 3) * BT;

  const float* fP  = features + ((size_t)bBase * E_ + a) * D_;
  const float* W1p = W1 + (size_t)(a * E_ + e) * D_ * H_;
  const float* b1p = b1 + (size_t)(a * E_ + e) * H_;
  const float* W2p = W2 + (size_t)(a * E_ + e) * H_ * D_;
  const float* b2p = b2 + (size_t)(a * E_ + e) * D_;

  // ---- Stage features tile [BT x D] -> bf16 LDS (row-major, K-contiguous) ----
#pragma unroll
  for (int pass = 0; pass < 16; ++pass) {
    const int r = pass * 2 + (tid >> 7);        // 0..31
    const int c = (tid & 127) * 4;              // 0..508
    const float4 v = *(const float4*)(fP + (size_t)r * (E_ * D_) + c);
    v4bf t;
    t.x = (bf16_t)v.x; t.y = (bf16_t)v.y;
    t.z = (bf16_t)v.z; t.w = (bf16_t)v.w;
    *(v4bf*)&featA[r * LDA1 + c] = t;           // one b64 store
  }
  __syncthreads();

  v8f acc2[8];                                  // [nt(4)][mt(2)] : 32x64 f32 tile
#pragma unroll
  for (int i = 0; i < 8; ++i) acc2[i] = (v8f){0, 0, 0, 0, 0, 0, 0, 0};

  const int nl  = lane & 15;
  const int mHi = (lane >> 4) * 8;

  for (int hc = 0; hc < H_ / HC; ++hc) {        // 16 H chunks
    const int hBase = hc * HC;

    // ================= GEMM1: h = features @ W1[:, hBase:hBase+HC] ==========
    v8f acc1[2];
    acc1[0] = (v8f){0, 0, 0, 0, 0, 0, 0, 0};
    acc1[1] = (v8f){0, 0, 0, 0, 0, 0, 0, 0};

    for (int kk = 0; kk < D_ / KSTAGE; ++kk) {  // K = 512, staged 64
      const int kBase = kk * KSTAGE;
      // stage W1[kBase..+64][hBase..+128] -> btB1[n][k] bf16, packed k-pairs
#pragma unroll
      for (int pass = 0; pass < 4; ++pass) {
        const int task = pass * NTHREADS + tid; // 0..1023
        const int rp = task >> 5;               // k row-pair 0..31 (rows 2rp,2rp+1)
        const int c  = (task & 31) * 4;         // h col 0..124
        const float4 v0 = *(const float4*)(W1p + (size_t)(kBase + 2 * rp)     * H_ + hBase + c);
        const float4 v1 = *(const float4*)(W1p + (size_t)(kBase + 2 * rp + 1) * H_ + hBase + c);
        store_pk(&btB1[(c + 0) * LDT + 2 * rp], v0.x, v1.x);
        store_pk(&btB1[(c + 1) * LDT + 2 * rp], v0.y, v1.y);
        store_pk(&btB1[(c + 2) * LDT + 2 * rp], v0.z, v1.z);
        store_pk(&btB1[(c + 3) * LDT + 2 * rp], v0.w, v1.w);
      }
      __syncthreads();

#pragma unroll
      for (int ks = 0; ks < KSTAGE / KC; ++ks) {
        const v16bf bfr = frag_B(btB1, wave * 16, ks * KC, LDT, lane);
#pragma unroll
        for (int mt = 0; mt < 2; ++mt) {
          const v16bf afr = frag_A(featA, mt * 16, kBase + ks * KC, LDA1, lane);
          acc1[mt] = wmma_bf16(afr, bfr, acc1[mt]);
        }
      }
      __syncthreads();
    }

    // ---- epilogue1: +b1, exact GELU, -> bf16 hA tile ----
    {
      const int n = wave * 16 + nl;             // h col within chunk
      const float b1v = b1p[hBase + n];
#pragma unroll
      for (int mt = 0; mt < 2; ++mt)
#pragma unroll
        for (int i = 0; i < 8; ++i) {
          const float x = acc1[mt][i] + b1v;
          const float g = 0.5f * x * (1.0f + erff(x * 0.70710678118654752f));
          hA[(mt * 16 + mHi + i) * LDA2 + n] = (bf16_t)g;
        }
    }
    __syncthreads();

    // ================= GEMM2 partial: acc2 += h @ W2[hBase:hBase+HC, :] =====
    for (int k2 = 0; k2 < HC / KSTAGE; ++k2) {  // 2 staged steps of K=64
      const int kB2 = k2 * KSTAGE;
      // stage W2[hBase+kB2..+64][0..512] -> btB2[d][k] bf16, packed k-pairs
#pragma unroll
      for (int pass = 0; pass < 16; ++pass) {
        const int task = pass * NTHREADS + tid; // 0..4095
        const int rp = task >> 7;               // k row-pair 0..31
        const int c  = (task & 127) * 4;        // d col 0..508
        const float4 v0 = *(const float4*)(W2p + (size_t)(hBase + kB2 + 2 * rp)     * D_ + c);
        const float4 v1 = *(const float4*)(W2p + (size_t)(hBase + kB2 + 2 * rp + 1) * D_ + c);
        store_pk(&btB2[(c + 0) * LDT + 2 * rp], v0.x, v1.x);
        store_pk(&btB2[(c + 1) * LDT + 2 * rp], v0.y, v1.y);
        store_pk(&btB2[(c + 2) * LDT + 2 * rp], v0.z, v1.z);
        store_pk(&btB2[(c + 3) * LDT + 2 * rp], v0.w, v1.w);
      }
      __syncthreads();

#pragma unroll
      for (int ks = 0; ks < KSTAGE / KC; ++ks) {
        v16bf aFr[2];
#pragma unroll
        for (int mt = 0; mt < 2; ++mt)
          aFr[mt] = frag_A(hA, mt * 16, kB2 + ks * KC, LDA2, lane);
#pragma unroll
        for (int nt = 0; nt < 4; ++nt) {
          const v16bf bFr = frag_B(btB2, wave * 64 + nt * 16, ks * KC, LDT, lane);
#pragma unroll
          for (int mt = 0; mt < 2; ++mt)
            acc2[nt * 2 + mt] = wmma_bf16(aFr[mt], bFr, acc2[nt * 2 + mt]);
        }
      }
      __syncthreads();
    }
  }

  // ================= epilogue2: +b2, squared error vs target, reduce ========
  float lsum = 0.0f;
#pragma unroll
  for (int nt = 0; nt < 4; ++nt) {
    const int d = wave * 64 + nt * 16 + nl;
    const float b2v = b2p[d];
#pragma unroll
    for (int mt = 0; mt < 2; ++mt)
#pragma unroll
      for (int i = 0; i < 8; ++i) {
        const int b = bBase + mt * 16 + mHi + i;
        const float err = acc2[nt * 2 + mt][i] + b2v -
                          target[((size_t)b * E_ + a) * D_ + d];
        lsum += err * err;
      }
  }
#pragma unroll
  for (int off = 16; off > 0; off >>= 1) lsum += __shfl_xor(lsum, off, 32);
  if (lane == 0) red[wave] = lsum;
  __syncthreads();
  if (tid == 0) {
    float s = 0.0f;
#pragma unroll
    for (int w = 0; w < NTHREADS / 32; ++w) s += red[w];
    atomicAdd(out, s * SCALE);
  }
}

extern "C" void kernel_launch(void* const* d_in, const int* in_sizes, int n_in,
                              void* d_out, int out_size, void* d_ws, size_t ws_size,
                              hipStream_t stream) {
  (void)in_sizes; (void)n_in; (void)out_size; (void)d_ws; (void)ws_size;
  const float* features = (const float*)d_in[0];
  const float* target   = (const float*)d_in[1];
  const float* W1       = (const float*)d_in[2];
  const float* b1       = (const float*)d_in[3];
  const float* W2       = (const float*)d_in[4];
  const float* b2       = (const float*)d_in[5];
  float* out = (float*)d_out;

  zero_out_kernel<<<dim3(1), dim3(1), 0, stream>>>(out);
  adapter_mse_kernel<<<dim3(256), dim3(NTHREADS), 0, stream>>>(
      features, target, W1, b1, W2, b2, out);
}
